// BiasMultiheadSelfAttention_43731357008134
// MI455X (gfx1250) — matrix operand order repaired and checked by reading, not compiled, not run on hardware
//
#include <hip/hip_runtime.h>
#include <hip/hip_bf16.h>

#define DEV __device__ __forceinline__

typedef __attribute__((ext_vector_type(16))) __bf16 v16bf;
typedef __attribute__((ext_vector_type(8)))  __bf16 v8bf;
typedef __attribute__((ext_vector_type(8)))  float  v8f;
typedef __attribute__((ext_vector_type(4)))  float  v4f;
typedef __attribute__((ext_vector_type(4)))  int    v4i;

constexpr int Bs = 16, Ns = 512, Ds = 768, Hs = 12, DHs = 64;

DEV v8f wmma_bf16(v16bf a, v16bf b, v8f c) {
  // D = A(16x32 bf16) * B(32x16 bf16) + C(16x16 f32)
  return __builtin_amdgcn_wmma_f32_16x16x32_bf16(false, a, false, b, (short)0, c,
                                                 false, false);
}

// ---- operand loaders (pure bf16) ------------------------------------------
// A-operand (16x32, M = lane&15): element j <-> K = 16*half + j  (contiguous 16)
// B-operand (32x16, N = lane&15): element j <-> K = 16*half + j  (contiguous 16)

DEV v16bf ldA_bf16(const __bf16* row, int kb, int half) {
  v8bf a = *(const v8bf*)(row + kb + 16 * half);
  v8bf b = *(const v8bf*)(row + kb + 16 * half + 8);
  v16bf r;
#pragma unroll
  for (int j = 0; j < 8; ++j) { r[j] = a[j]; r[j + 8] = b[j]; }
  return r;
}

DEV v16bf ldB_bf16(const __bf16* row, int kb, int half) {
  return *(const v16bf*)(row + kb + 16 * half);
}

// ---- prep: one-time f32 -> bf16 conversion --------------------------------
__global__ __launch_bounds__(256) void cvt_bf16(const float* __restrict__ in,
                                                __bf16* __restrict__ out, int n8) {
  const int i = blockIdx.x * 256 + threadIdx.x;
  if (i >= n8) return;
  const v4f a = *(const v4f*)(in + i * 8);
  const v4f b = *(const v4f*)(in + i * 8 + 4);
  v8bf o;
#pragma unroll
  for (int j = 0; j < 4; ++j) { o[j] = (__bf16)a[j]; o[j + 4] = (__bf16)b[j]; }
  *(v8bf*)(out + i * 8) = o;
}

// ---- kernel 1: fused QKV projection (transposed: D = W * x^T) -------------
// wave computes 16(j) x 64(m); lane holds one activation row m -> vector stores
__global__ __launch_bounds__(256) void qkv_proj(
    const __bf16* __restrict__ xb, const __bf16* __restrict__ wb,
    const float* __restrict__ bias, __bf16* __restrict__ qo,
    __bf16* __restrict__ ko, __bf16* __restrict__ vT) {
  const int wave = (blockIdx.x * 256 + threadIdx.x) >> 5;
  const int lane = threadIdx.x & 31;
  const int half = lane >> 4, idx = lane & 15;
  const int jtile = wave / 128;                 // 144 j-tiles of 16 (3D/16)
  const int mcol = (wave - jtile * 128) * 64;   // 128 m-cols of 64 (B*N/64)
  const __bf16* wrow = wb + (jtile * 16 + idx) * Ds;

  v8f acc[4] = {};
  for (int kb = 0; kb < Ds; kb += 32) {
    v16bf a = ldA_bf16(wrow, kb, half);
#pragma unroll
    for (int t = 0; t < 4; ++t)
      acc[t] = wmma_bf16(a, ldB_bf16(xb + (mcol + t * 16 + idx) * Ds, kb, half),
                         acc[t]);
  }

  const int jbase = jtile * 16;
  const int blk = jbase / Ds;  // 0=Q 1=K 2=V (uniform per wave)
  const int jj = jbase - blk * Ds;
  const int h = jj >> 6;
  const int dhb = (jj & 63) + 8 * half;  // dh of VGPR slot r=0
  const float* bp = bias + jbase + 8 * half;
  const v4f bj0 = *(const v4f*)bp, bj1 = *(const v4f*)(bp + 4);
  const float scl = (blk == 0) ? 0.125f : 1.0f;

#pragma unroll
  for (int t = 0; t < 4; ++t) {
    const int m = mcol + t * 16 + idx;
    const int bidx = m >> 9, n = m & 511;
    if (blk == 2) {
#pragma unroll
      for (int r = 0; r < 8; ++r) {
        const float bj = (r < 4) ? bj0[r] : bj1[r - 4];
        vT[(((bidx * Hs + h) * DHs + dhb + r) << 9) + n] = (__bf16)(acc[t][r] + bj);
      }
    } else {
      v8bf o;
#pragma unroll
      for (int r = 0; r < 8; ++r) {
        const float bj = (r < 4) ? bj0[r] : bj1[r - 4];
        o[r] = (__bf16)((acc[t][r] + bj) * scl);
      }
      __bf16* dst = ((blk == 0) ? qo : ko) + (((bidx * Hs + h) * Ns + n) << 6) + dhb;
      *(v8bf*)dst = o;  // contiguous dh, 16B aligned
    }
  }
}

// ---- kernel 2: flash attention (transposed scores) ------------------------
// S^T = K * Q^T  =>  C-layout: key = r + 8*half, query = lane&15 (per-lane stats)
__global__ __launch_bounds__(256) void attn(
    const __bf16* __restrict__ q, const __bf16* __restrict__ k,
    const __bf16* __restrict__ vT, const float* __restrict__ bias,
    const int* __restrict__ kpm, __bf16* __restrict__ ctx) {
  const int wave = (blockIdx.x * 256 + threadIdx.x) >> 5;
  const int lane = threadIdx.x & 31;
  const int half = lane >> 4, idx = lane & 15;
  const int qtile = wave & 31;  // 32 query tiles
  const int bh = wave >> 5;     // 0..191
  const int b = bh / Hs;
  const int h = bh - b * Hs;

  const __bf16* kp = k + bh * (Ns * DHs);
  const __bf16* vp = vT + bh * (DHs * Ns);
  const int qglob = qtile * 16 + idx;
  const __bf16* qrow = q + bh * (Ns * DHs) + qglob * DHs;
  const float* brow = bias + (size_t)bh * Ns * Ns + (size_t)qglob * Ns;
  const int* mp = kpm + b * Ns;

  const v16bf Qb0 = ldB_bf16(qrow, 0, half);   // dh 0..31 (B operand)
  const v16bf Qb1 = ldB_bf16(qrow, 32, half);  // dh 32..63

  v8f C[4] = {};
  float row_max = -1e30f, row_sum = 0.f;

  for (int kt = 0; kt < Ns; kt += 32) {
    if (kt + 32 < Ns) __builtin_prefetch(brow + kt + 32 + 8 * half, 0, 3);
    float p[2][8];
    float mloc = -1e30f;
#pragma unroll
    for (int hk = 0; hk < 2; ++hk) {  // two 16-key half tiles
      const int kbb = kt + hk * 16;
      const __bf16* krow = kp + (kbb + idx) * DHs;
      v8f S = {};
      S = wmma_bf16(ldA_bf16(krow, 0, half), Qb0, S);
      S = wmma_bf16(ldA_bf16(krow, 32, half), Qb1, S);
      const float* bv = brow + kbb + 8 * half;
      const int* mv = mp + kbb + 8 * half;
      v4f b0 = *(const v4f*)bv, b1 = *(const v4f*)(bv + 4);
      v4i m0 = *(const v4i*)mv, m1 = *(const v4i*)(mv + 4);
#pragma unroll
      for (int r = 0; r < 8; ++r) {
        const float bb = (r < 4) ? b0[r] : b1[r - 4];
        const int mm = (r < 4) ? m0[r] : m1[r - 4];
        const float s = S[r] + bb + (mm ? -10000.f : 0.f);
        p[hk][r] = s;
        mloc = fmaxf(mloc, s);
      }
    }
    // per-query tile max: combine the two 16-key halves (lanes l <-> l+16)
    const float mt = fmaxf(mloc, __shfl_xor(mloc, 16, 32));
    const float nmax = fmaxf(row_max, mt);
    const float corr = __expf(row_max - nmax);
    float sloc = 0.f;
#pragma unroll
    for (int hk = 0; hk < 2; ++hk)
#pragma unroll
      for (int r = 0; r < 8; ++r) {
        p[hk][r] = __expf(p[hk][r] - nmax);
        sloc += p[hk][r];
      }
    row_sum = row_sum * corr + sloc + __shfl_xor(sloc, 16, 32);
    row_max = nmax;

    // Build P^T as B-operand: lane needs keys 16*half + 0..15.
    // Own keys: 8*half+r (hk0) and 16+8*half+r (hk1); swap one group with partner.
    v16bf P;
#pragma unroll
    for (int r = 0; r < 8; ++r) {
      const float send = half ? p[0][r] : p[1][r];
      const float recv = __shfl_xor(send, 16, 32);
      const float lo = half ? recv : p[0][r];  // j=r   -> key 16*half + r
      const float hi = half ? p[1][r] : recv;  // j=8+r -> key 16*half + 8 + r
      P[r] = (__bf16)lo;
      P[r + 8] = (__bf16)hi;
    }
    // ctx^T += V^T * P^T (4 dh-chunks of 16), rescale by corr first
#pragma unroll
    for (int c = 0; c < 4; ++c) {
#pragma unroll
      for (int r = 0; r < 8; ++r) C[c][r] *= corr;
      C[c] = wmma_bf16(ldA_bf16(vp + (c * 16 + idx) * Ns, kt, half), P, C[c]);
    }
  }

  const float inv = 1.f / row_sum;
  __bf16* crow = ctx + (size_t)(b * Ns + qglob) * Ds + h * DHs;
#pragma unroll
  for (int c = 0; c < 4; ++c) {
    v8bf o;
#pragma unroll
    for (int r = 0; r < 8; ++r) o[r] = (__bf16)(C[c][r] * inv);
    *(v8bf*)(crow + c * 16 + 8 * half) = o;  // dh = c*16 + 8*half + r
  }
}

// ---- kernel 3: output projection (transposed: D = W * ctx^T) --------------
// lane holds one row m -> contiguous fp32 vector stores
__global__ __launch_bounds__(256) void out_proj(
    const __bf16* __restrict__ ctx, const __bf16* __restrict__ wb,
    const float* __restrict__ bias, float* __restrict__ out) {
  const int wave = (blockIdx.x * 256 + threadIdx.x) >> 5;
  const int lane = threadIdx.x & 31;
  const int half = lane >> 4, idx = lane & 15;
  const int jtile = wave / 128;                // 48 j-tiles of 16
  const int mcol = (wave - jtile * 128) * 64;  // 128 m-cols of 64
  const __bf16* wrow = wb + (jtile * 16 + idx) * Ds;

  v8f acc[4] = {};
  for (int kb = 0; kb < Ds; kb += 32) {
    v16bf a = ldA_bf16(wrow, kb, half);
#pragma unroll
    for (int t = 0; t < 4; ++t)
      acc[t] = wmma_bf16(a, ldB_bf16(ctx + (mcol + t * 16 + idx) * Ds, kb, half),
                         acc[t]);
  }

  const int jbase = jtile * 16 + 8 * half;
  const v4f bj0 = *(const v4f*)(bias + jbase);
  const v4f bj1 = *(const v4f*)(bias + jbase + 4);
#pragma unroll
  for (int t = 0; t < 4; ++t) {
    const int m = mcol + t * 16 + idx;
    v4f o0, o1;
#pragma unroll
    for (int r = 0; r < 4; ++r) {
      o0[r] = acc[t][r] + bj0[r];
      o1[r] = acc[t][r + 4] + bj1[r];
    }
    float* dst = out + m * Ds + jbase;
    *(v4f*)dst = o0;
    *(v4f*)(dst + 4) = o1;
  }
}

extern "C" void kernel_launch(void* const* d_in, const int* in_sizes, int n_in,
                              void* d_out, int out_size, void* d_ws,
                              size_t ws_size, hipStream_t stream) {
  (void)in_sizes; (void)n_in; (void)out_size; (void)ws_size;
  const float* x = (const float*)d_in[0];
  const int* kpm = (const int*)d_in[1];
  const float* attn_bias = (const float*)d_in[2];
  const float* w_in = (const float*)d_in[3];
  const float* b_in = (const float*)d_in[4];
  const float* w_out = (const float*)d_in[5];
  const float* b_out = (const float*)d_in[6];

  char* ws = (char*)d_ws;
  const size_t seg = (size_t)Bs * Hs * Ns * DHs * sizeof(__bf16);  // 12.58 MB
  __bf16* qb = (__bf16*)(ws);
  __bf16* kb = (__bf16*)(ws + seg);
  __bf16* vT = (__bf16*)(ws + 2 * seg);
  __bf16* ctx = (__bf16*)(ws + 3 * seg);
  __bf16* xb = (__bf16*)(ws + 4 * seg);
  __bf16* winb = (__bf16*)(ws + 5 * seg);                      // 3.54 MB
  __bf16* woutb = (__bf16*)(ws + 5 * seg + (size_t)3 * Ds * Ds * 2);  // 1.18 MB

  // one-time f32 -> bf16 operand conversion (all fit in L2 afterwards)
  const int nx8 = Bs * Ns * Ds / 8;       // 786432
  const int nwi8 = 3 * Ds * Ds / 8;       // 221184
  const int nwo8 = Ds * Ds / 8;           // 73728
  cvt_bf16<<<(nx8 + 255) / 256, 256, 0, stream>>>(x, xb, nx8);
  cvt_bf16<<<(nwi8 + 255) / 256, 256, 0, stream>>>(w_in, winb, nwi8);
  cvt_bf16<<<(nwo8 + 255) / 256, 256, 0, stream>>>(w_out, woutb, nwo8);

  // 144 j-tiles * 128 m-cols = 18432 waves -> 2304 blocks of 8 waves
  qkv_proj<<<2304, 256, 0, stream>>>(xb, winb, b_in, qb, kb, vT);
  // 192 (b,h) * 32 q-tiles = 6144 waves -> 768 blocks
  attn<<<768, 256, 0, stream>>>(qb, kb, vT, attn_bias, kpm, ctx);
  // 48 j-tiles * 128 m-cols = 6144 waves -> 768 blocks
  out_proj<<<768, 256, 0, stream>>>(ctx, woutb, b_out, (float*)d_out);
}
